// StatefulMambaTextEncoder_84722524881493
// MI455X (gfx1250) — compile-verified, matrix operand order checked by reading, not run on hardware
//
#include <hip/hip_runtime.h>
#include <cstdint>

// ---------------------------------------------------------------------------
// Problem constants
// ---------------------------------------------------------------------------
#define Bsz   32
#define Tlen  2000
#define Edim  256
#define Sdim  64
#define Lnum  3
#define NTOK  (Bsz * Tlen)          // 64000 token rows
#define NC    20                    // scan chunks
#define CL    100                   // chunk length (NC*CL == Tlen)

// ---------------------------------------------------------------------------
// CDNA5 WMMA types
// ---------------------------------------------------------------------------
typedef __attribute__((ext_vector_type(16))) __bf16     v16bf;
typedef __attribute__((ext_vector_type(8)))  float      v8f;
typedef __attribute__((ext_vector_type(4)))  unsigned   u32x4;

union FragBF { u32x4 u[2]; v16bf v; };

__device__ __forceinline__ v8f zero_v8f() {
    v8f z;
#pragma unroll
    for (int i = 0; i < 8; ++i) z[i] = 0.0f;
    return z;
}

__device__ __forceinline__ v16bf zero_frag() {
    FragBF f;
    f.u[0] = (u32x4)0u;
    f.u[1] = (u32x4)0u;
    return f.v;
}

// Load one 16x32 bf16 A/B fragment row for this lane.
// Lane layout per ISA 7.12.2: lanes 0..15 hold K{kcol+0..7, kcol+16..23},
// lanes 16..31 hold K{kcol+8..15, kcol+24..31}.
__device__ __forceinline__ v16bf load_frag(const __bf16* rowptr, int kcol, int hf) {
    FragBF f;
    f.u[0] = *(const u32x4*)(rowptr + kcol + hf * 8);
    f.u[1] = *(const u32x4*)(rowptr + kcol + 16 + hf * 8);
    return f.v;
}

__device__ __forceinline__ v8f wmma_bf16(v16bf a, v16bf b, v8f c) {
    return __builtin_amdgcn_wmma_f32_16x16x32_bf16(
        false, a, false, b, (short)0, c, false, false);
}

__device__ __forceinline__ unsigned short f2bf(float f) {
    unsigned u = __float_as_uint(f);
    unsigned r = (u + 0x7FFFu + ((u >> 16) & 1u)) >> 16;
    return (unsigned short)r;
}

__device__ __forceinline__ float silu_f(float x) {
    return x / (1.0f + __expf(-x));
}

// ---------------------------------------------------------------------------
// Small helper kernels
// ---------------------------------------------------------------------------
__global__ void cvt_f32_bf16(const float* __restrict__ src,
                             unsigned short* __restrict__ dst, int n) {
    int i = blockIdx.x * 256 + threadIdx.x;
    if (i < n) dst[i] = f2bf(src[i]);
}

// ctx_w (4, 64, 256, 3) -> bf16 (4, 64, 3*256) with K = tap*256 + cin
__global__ void ctx_w_transpose(const float* __restrict__ src,
                                unsigned short* __restrict__ dst) {
    int i = blockIdx.x * 256 + threadIdx.x;   // 4*64*768 = 196608 total
    if (i >= 4 * 64 * 768) return;
    int c    = i % 768;
    int oc   = (i / 768) % 64;
    int j    = i / (768 * 64);
    int ktap = c >> 8;
    int cin  = c & 255;
    dst[i] = f2bf(src[((j * 64 + oc) * 256 + cin) * 3 + ktap]);
}

// x[b,t,e] = tok_emb[tokens[b,t], e] + pos_emb[t, e]
__global__ void embed_kernel(const int* __restrict__ tokens,
                             const float* __restrict__ tok_emb,
                             const float* __restrict__ pos_emb,
                             float* __restrict__ x,
                             unsigned short* __restrict__ xbf) {
    int gt = blockIdx.x;
    int e  = threadIdx.x;
    int t  = gt % Tlen;
    int tk = tokens[gt];
    float v = tok_emb[tk * Edim + e] + pos_emb[t * Edim + e];
    long idx = (long)gt * Edim + e;
    x[idx]   = v;
    xbf[idx] = f2bf(v);
}

// Depthwise causal conv (kernel 4, left pad 3) + SiLU
__global__ void dwconv_silu(const float* __restrict__ xin,
                            const float* __restrict__ cw,
                            const float* __restrict__ cb,
                            float* __restrict__ xc,
                            unsigned short* __restrict__ xcbf) {
    int gt = blockIdx.x;
    int e  = threadIdx.x;
    int tl = gt % Tlen;
    float acc = cb[e];
#pragma unroll
    for (int k = 0; k < 4; ++k) {
        int ts = tl - 3 + k;
        if (ts >= 0) acc += xin[(long)(gt - 3 + k) * Edim + e] * cw[e * 4 + k];
    }
    float y = silu_f(acc);
    long idx = (long)gt * Edim + e;
    xc[idx]   = y;
    xcbf[idx] = f2bf(y);
}

// ---------------------------------------------------------------------------
// Generic WMMA GEMM: out[m,n] = sum_k A[m,k] * W[n,k]   (both K-contiguous)
// Block = 256 threads = 8 waves arranged 4(M) x 2(N); wave tile 32x64.
// Block tile 128x128.  M%128==0, N%128==0, K%32==0 guaranteed.
// Per k-step: 6 b128-pair fragment loads feed 8 v_wmma (0.75 loads/wmma).
// ---------------------------------------------------------------------------
#define MODE_PLAIN      0
#define MODE_SPLIT_SILU 1
#define MODE_BIAS_RELU  2

__global__ __launch_bounds__(256) void gemm_bf16_wmma(
    const __bf16* __restrict__ A, const __bf16* __restrict__ W,
    int K, int mode, const float* __restrict__ bias,
    float* __restrict__ out0, float* __restrict__ out1, int ldo) {

    const int wave = threadIdx.x >> 5;
    const int lane = threadIdx.x & 31;
    const int wm   = wave >> 1;          // 0..3 : 32 rows each
    const int wn   = wave & 1;           // 0..1 : 64 cols each
    const int hf   = lane >> 4;          // fragment half
    const int r16  = lane & 15;

    const int blockM = blockIdx.x * 128;
    const int blockN = blockIdx.y * 128;

    const __bf16* a0p = A + (long)(blockM + wm * 32 + r16) * K;
    const __bf16* a1p = a0p + (long)16 * K;
    const __bf16* bp[4];
#pragma unroll
    for (int j = 0; j < 4; ++j)
        bp[j] = W + (long)(blockN + wn * 64 + j * 16 + r16) * K;

    v8f acc[2][4];
#pragma unroll
    for (int i = 0; i < 2; ++i)
#pragma unroll
        for (int j = 0; j < 4; ++j) acc[i][j] = zero_v8f();

    for (int k = 0; k < K; k += 32) {
        __builtin_prefetch(a0p + k + 128, 0, 1);   // global_prefetch_b8
        v16bf a0 = load_frag(a0p, k, hf);
        v16bf a1 = load_frag(a1p, k, hf);
        v16bf b0 = load_frag(bp[0], k, hf);
        v16bf b1 = load_frag(bp[1], k, hf);
        v16bf b2 = load_frag(bp[2], k, hf);
        v16bf b3 = load_frag(bp[3], k, hf);
        acc[0][0] = wmma_bf16(a0, b0, acc[0][0]);
        acc[0][1] = wmma_bf16(a0, b1, acc[0][1]);
        acc[0][2] = wmma_bf16(a0, b2, acc[0][2]);
        acc[0][3] = wmma_bf16(a0, b3, acc[0][3]);
        acc[1][0] = wmma_bf16(a1, b0, acc[1][0]);
        acc[1][1] = wmma_bf16(a1, b1, acc[1][1]);
        acc[1][2] = wmma_bf16(a1, b2, acc[1][2]);
        acc[1][3] = wmma_bf16(a1, b3, acc[1][3]);
    }

#pragma unroll
    for (int i = 0; i < 2; ++i)
#pragma unroll
        for (int j = 0; j < 4; ++j)
#pragma unroll
            for (int r = 0; r < 8; ++r) {
                int row = blockM + wm * 32 + i * 16 + hf * 8 + r;
                int col = blockN + wn * 64 + j * 16 + r16;
                float v = acc[i][j][r];
                if (mode == MODE_SPLIT_SILU) {
                    if (col < Edim) out0[(long)row * Edim + col] = v;
                    else            out1[(long)row * Edim + (col - Edim)] = silu_f(v);
                } else if (mode == MODE_BIAS_RELU) {
                    v += bias[col];
                    out0[(long)row * ldo + col] = fmaxf(v, 0.0f);
                } else {
                    out0[(long)row * ldo + col] = v;
                }
            }
}

// ---------------------------------------------------------------------------
// Chunk-parallel SSM scan.  dA is time-invariant, so the diagonal recurrence
//   h_t = dA*h_{t-1} + (dt*u_t)*B_t    splits into 3 passes:
//   pass1: per chunk, local scan with h0=0 -> end state   (B only)
//   pass2: per (b,d,s), cross-chunk recurrence with dA^CL  (NC steps)
//   pass3: per chunk, replay with corrected start state, emit gated output
// State layout [b][c][s][d] -> coalesced across d everywhere.
// ---------------------------------------------------------------------------
__global__ __launch_bounds__(256) void scan_pass1(
    const float* __restrict__ xc, const float* __restrict__ bc,
    const float* __restrict__ dt_w, const float* __restrict__ A_log,
    float* __restrict__ hloc) {

    const int b = blockIdx.x / NC;
    const int c = blockIdx.x % NC;
    const int d = threadIdx.x;

    const float dt = logf(1.0f + __expf(dt_w[d]));
    float dA[Sdim], h[Sdim];
#pragma unroll
    for (int s = 0; s < Sdim; ++s) {
        dA[s] = 1.0f - dt * __expf(A_log[d * Sdim + s]);
        h[s]  = 0.0f;
    }

    __shared__ float Bs[Sdim];
    for (int t = c * CL; t < (c + 1) * CL; ++t) {
        long row = (long)b * Tlen + t;
        if (d < Sdim) Bs[d] = bc[row * (2 * Sdim) + d];
        __syncthreads();
        float du = dt * xc[row * Edim + d];
#pragma unroll
        for (int s = 0; s < Sdim; ++s) h[s] = dA[s] * h[s] + du * Bs[s];
        __syncthreads();
    }
    long base = ((long)(b * NC + c) * Sdim) * Edim + d;
#pragma unroll
    for (int s = 0; s < Sdim; ++s) hloc[base + (long)s * Edim] = h[s];
}

__global__ __launch_bounds__(256) void scan_pass2(
    const float* __restrict__ hloc,
    const float* __restrict__ dt_w, const float* __restrict__ A_log,
    float* __restrict__ hstart) {

    int idx = blockIdx.x * 256 + threadIdx.x;        // (b*64 + s)*256 + d
    int d = idx & 255;
    int s = (idx >> 8) & (Sdim - 1);
    int b = idx >> 14;

    float dt = logf(1.0f + __expf(dt_w[d]));
    float a  = 1.0f - dt * __expf(A_log[d * Sdim + s]);
    float aP = __powf(a, (float)CL);

    float hs = 0.0f;
    long stride = (long)Sdim * Edim;
    long base   = ((long)b * NC * Sdim + s) * Edim + d;
    hstart[base] = 0.0f;
    for (int c = 1; c < NC; ++c) {
        hs = aP * hs + hloc[base + (long)(c - 1) * stride];
        hstart[base + (long)c * stride] = hs;
    }
}

__global__ __launch_bounds__(256) void scan_pass3(
    const float* __restrict__ xc, const float* __restrict__ bc,
    const float* __restrict__ sres, const float* __restrict__ hstart,
    const float* __restrict__ dt_w, const float* __restrict__ A_log,
    const float* __restrict__ D_param,
    unsigned short* __restrict__ g_bf) {

    const int b = blockIdx.x / NC;
    const int c = blockIdx.x % NC;
    const int d = threadIdx.x;

    const float dt = logf(1.0f + __expf(dt_w[d]));
    const float Dp = D_param[d];
    float dA[Sdim], h[Sdim];
    long hbase = ((long)(b * NC + c) * Sdim) * Edim + d;
#pragma unroll
    for (int s = 0; s < Sdim; ++s) {
        dA[s] = 1.0f - dt * __expf(A_log[d * Sdim + s]);
        h[s]  = hstart[hbase + (long)s * Edim];
    }

    __shared__ float Bs[Sdim];
    __shared__ float Cs[Sdim];
    for (int t = c * CL; t < (c + 1) * CL; ++t) {
        long row = (long)b * Tlen + t;
        if (d < 2 * Sdim) {
            float v = bc[row * (2 * Sdim) + d];
            if (d < Sdim) Bs[d] = v; else Cs[d - Sdim] = v;
        }
        __syncthreads();
        float u  = xc[row * Edim + d];
        float du = dt * u;
        float y  = 0.0f;
#pragma unroll
        for (int s = 0; s < Sdim; ++s) {
            h[s] = dA[s] * h[s] + du * Bs[s];
            y += Cs[s] * h[s];
        }
        float g = (y + Dp * u) * sres[row * Edim + d];
        g_bf[row * Edim + d] = f2bf(g);
        __syncthreads();
    }
}

// ---------------------------------------------------------------------------
// Residual + LayerNorm (block per token) + optional strided bf16 copy.
// ---------------------------------------------------------------------------
__global__ __launch_bounds__(256) void ln_kernel(
    const float* __restrict__ yin, const float* __restrict__ resid,
    const float* __restrict__ gamma, const float* __restrict__ beta,
    float* __restrict__ fout, unsigned short* __restrict__ bfout,
    int bf_ld, int bf_off) {

    int gt = blockIdx.x;
    int e  = threadIdx.x;
    __shared__ float buf[Edim];

    float v = yin[(long)gt * Edim + e];
    if (resid) v += resid[(long)gt * Edim + e];

    buf[e] = v;
    __syncthreads();
    for (int off = 128; off > 0; off >>= 1) {
        if (e < off) buf[e] += buf[e + off];
        __syncthreads();
    }
    float mean = buf[0] * (1.0f / Edim);
    __syncthreads();

    float dv = v - mean;
    buf[e] = dv * dv;
    __syncthreads();
    for (int off = 128; off > 0; off >>= 1) {
        if (e < off) buf[e] += buf[e + off];
        __syncthreads();
    }
    float var = buf[0] * (1.0f / Edim);

    float o = dv * rsqrtf(var + 1e-5f) * gamma[e] + beta[e];
    if (fout)  fout[(long)gt * Edim + e] = o;
    if (bfout) bfout[(long)gt * bf_ld + bf_off + e] = f2bf(o);
}

// ---------------------------------------------------------------------------
// Dilated context conv as WMMA GEMM over K = 3*256 with shifted A rows.
// ---------------------------------------------------------------------------
__global__ __launch_bounds__(256) void ctx_wmma_kernel(
    const __bf16* __restrict__ xbf, const __bf16* __restrict__ wctx,
    const float* __restrict__ ctx_b, unsigned short* __restrict__ cat) {

    const int j  = blockIdx.y;
    const int d  = 1 << j;                 // dilation 1,2,4,8
    const int wave = threadIdx.x >> 5;
    const int lane = threadIdx.x & 31;
    const int wm = wave >> 1;              // 0..3  (16 rows each)
    const int wn = wave & 1;               // 0..1  (32 cols each)
    const int hf  = lane >> 4;
    const int r16 = lane & 15;

    const int m  = blockIdx.x * 64 + wm * 16 + r16;  // this lane's token row
    const int tl = m % Tlen;

    const __bf16* Wj  = wctx + (long)j * 64 * 768;
    const __bf16* b0p = Wj + (long)(wn * 32 + r16) * 768;
    const __bf16* b1p = b0p + (long)16 * 768;

    v8f acc0 = zero_v8f();
    v8f acc1 = zero_v8f();

    for (int kk = 0; kk < 24; ++kk) {
        int k    = kk * 32;
        int ktap = k >> 8;                 // 0,1,2 -> time offsets -d,0,+d
        int kc   = k & 255;
        int ts   = tl + (ktap - 1) * d;
        v16bf a = (ts >= 0 && ts < Tlen)
                    ? load_frag(xbf + (long)(m + (ktap - 1) * d) * Edim, kc, hf)
                    : zero_frag();
        v16bf b0 = load_frag(b0p, k, hf);
        v16bf b1 = load_frag(b1p, k, hf);
        acc0 = wmma_bf16(a, b0, acc0);
        acc1 = wmma_bf16(a, b1, acc1);
    }

#pragma unroll
    for (int jt = 0; jt < 2; ++jt) {
        v8f acc = jt ? acc1 : acc0;
#pragma unroll
        for (int r = 0; r < 8; ++r) {
            int row = blockIdx.x * 64 + wm * 16 + hf * 8 + r;
            int ch  = wn * 32 + jt * 16 + r16;             // 0..63
            float v = acc[r] + ctx_b[j * 64 + ch];
            cat[(long)row * 512 + Edim + j * 64 + ch] = f2bf(v);
        }
    }
}

// ---------------------------------------------------------------------------
// Host-side orchestration
// ---------------------------------------------------------------------------
extern "C" void kernel_launch(void* const* d_in, const int* in_sizes, int n_in,
                              void* d_out, int out_size, void* d_ws, size_t ws_size,
                              hipStream_t stream) {
    (void)in_sizes; (void)n_in; (void)out_size; (void)ws_size;

    const int*   tokens   = (const int*)  d_in[0];
    const float* tok_emb  = (const float*)d_in[1];
    const float* pos_emb  = (const float*)d_in[2];
    const float* in_w     = (const float*)d_in[3];
    const float* conv_w   = (const float*)d_in[4];
    const float* conv_b   = (const float*)d_in[5];
    const float* dt_w     = (const float*)d_in[6];
    const float* A_log    = (const float*)d_in[7];
    const float* D_param  = (const float*)d_in[8];
    const float* B_w      = (const float*)d_in[9];
    const float* C_w      = (const float*)d_in[10];
    const float* out_w    = (const float*)d_in[11];
    const float* ln_g     = (const float*)d_in[12];
    const float* ln_b     = (const float*)d_in[13];
    const float* ctx_w    = (const float*)d_in[14];
    const float* ctx_b    = (const float*)d_in[15];
    const float* fus_w    = (const float*)d_in[16];
    const float* fus_b    = (const float*)d_in[17];
    const float* fus_ln_g = (const float*)d_in[18];
    const float* fus_ln_b = (const float*)d_in[19];

    // ---- workspace layout (bytes) ----
    const size_t F32_E  = (size_t)NTOK * Edim * 4;                 // 65.5 MB
    const size_t BF_E   = (size_t)NTOK * Edim * 2;                 // 32.8 MB
    const size_t BF_2E  = (size_t)NTOK * 512  * 2;                 // 65.5 MB
    const size_t F32_BC = (size_t)NTOK * 128  * 4;                 // 32.8 MB
    const size_t F32_H  = (size_t)Bsz * NC * Sdim * Edim * 4;      // 41.9 MB

    char* ws = (char*)d_ws;
    size_t off = 0;
    float*          x      = (float*)(ws + off); off += F32_E;
    float*          feats  = (float*)(ws + off); off += F32_E;
    float*          x_in   = (float*)(ws + off); off += F32_E;
    float*          sresb  = (float*)(ws + off); off += F32_E;
    float*          xc     = (float*)(ws + off); off += F32_E;
    float*          yo     = (float*)(ws + off); off += F32_E;
    float*          bc     = (float*)(ws + off); off += F32_BC;
    float*          hloc   = (float*)(ws + off); off += F32_H;
    float*          hstart = (float*)(ws + off); off += F32_H;
    unsigned short* x_bf   = (unsigned short*)(ws + off); off += BF_E;
    unsigned short* a_bf   = (unsigned short*)(ws + off); off += BF_E;
    unsigned short* xc_bf  = (unsigned short*)(ws + off); off += BF_E;
    unsigned short* g_bf   = (unsigned short*)(ws + off); off += BF_E;
    unsigned short* cat_bf = (unsigned short*)(ws + off); off += BF_2E;
    unsigned short* iw_bf   = (unsigned short*)(ws + off); off += (size_t)Lnum * 512 * Edim * 2;
    unsigned short* bcw_bf  = (unsigned short*)(ws + off); off += (size_t)Lnum * 128 * Edim * 2;
    unsigned short* ow_bf   = (unsigned short*)(ws + off); off += (size_t)Lnum * Edim * Edim * 2;
    unsigned short* fusw_bf = (unsigned short*)(ws + off); off += (size_t)Edim * 512 * 2;
    unsigned short* ctxw_bf = (unsigned short*)(ws + off); off += (size_t)4 * 64 * 768 * 2;

    auto cvt = [&](const float* s, unsigned short* dst, int n) {
        cvt_f32_bf16<<<(n + 255) / 256, 256, 0, stream>>>(s, dst, n);
    };

    // ---- weight conversion ----
    cvt(in_w,  iw_bf,  Lnum * 512 * Edim);
    for (int l = 0; l < Lnum; ++l) {
        cvt(B_w + (size_t)l * Sdim * Edim, bcw_bf + (size_t)l * 128 * Edim,               Sdim * Edim);
        cvt(C_w + (size_t)l * Sdim * Edim, bcw_bf + (size_t)l * 128 * Edim + Sdim * Edim, Sdim * Edim);
    }
    cvt(out_w, ow_bf,  Lnum * Edim * Edim);
    cvt(fus_w, fusw_bf, Edim * 512);
    ctx_w_transpose<<<(4 * 64 * 768 + 255) / 256, 256, 0, stream>>>(ctx_w, ctxw_bf);

    // ---- embedding ----
    embed_kernel<<<NTOK, 256, 0, stream>>>(tokens, tok_emb, pos_emb, x, x_bf);

    // ---- SSM layers ----
    for (int l = 0; l < Lnum; ++l) {
        const __bf16* Abf = (const __bf16*)(l == 0 ? x_bf : a_bf);
        const float* dtl = dt_w    + (size_t)l * Edim;
        const float* All = A_log   + (size_t)l * Edim * Sdim;
        const float* Dl  = D_param + (size_t)l * Edim;

        // in-proj: N=512, split epilogue (x_in | silu(x_res))
        gemm_bf16_wmma<<<dim3(NTOK / 128, 512 / 128), 256, 0, stream>>>(
            Abf, (const __bf16*)(iw_bf + (size_t)l * 512 * Edim),
            Edim, MODE_SPLIT_SILU, nullptr, x_in, sresb, Edim);

        dwconv_silu<<<NTOK, 256, 0, stream>>>(
            x_in, conv_w + (size_t)l * Edim * 4, conv_b + (size_t)l * Edim, xc, xc_bf);

        // B/C projection: N=128 (B rows then C rows)
        gemm_bf16_wmma<<<dim3(NTOK / 128, 1), 256, 0, stream>>>(
            (const __bf16*)xc_bf, (const __bf16*)(bcw_bf + (size_t)l * 128 * Edim),
            Edim, MODE_PLAIN, nullptr, bc, nullptr, 128);

        // chunk-parallel scan
        scan_pass1<<<Bsz * NC, 256, 0, stream>>>(xc, bc, dtl, All, hloc);
        scan_pass2<<<(Bsz * Sdim * Edim) / 256, 256, 0, stream>>>(hloc, dtl, All, hstart);
        scan_pass3<<<Bsz * NC, 256, 0, stream>>>(xc, bc, sresb, hstart, dtl, All, Dl, g_bf);

        // out-proj: N=256
        gemm_bf16_wmma<<<dim3(NTOK / 128, Edim / 128), 256, 0, stream>>>(
            (const __bf16*)g_bf, (const __bf16*)(ow_bf + (size_t)l * Edim * Edim),
            Edim, MODE_PLAIN, nullptr, yo, nullptr, Edim);

        // residual + LN; bf16 copy feeds next GEMM (or concat buffer, layer 2)
        ln_kernel<<<NTOK, 256, 0, stream>>>(
            yo, l == 0 ? x : feats,
            ln_g + (size_t)l * Edim, ln_b + (size_t)l * Edim,
            feats,
            l == Lnum - 1 ? cat_bf : a_bf,
            l == Lnum - 1 ? 512 : Edim, 0);
    }

    // ---- dilated context convs -> concat columns [256, 512) ----
    ctx_wmma_kernel<<<dim3(NTOK / 64, 4), 256, 0, stream>>>(
        (const __bf16*)x_bf, (const __bf16*)ctxw_bf, ctx_b, cat_bf);

    // ---- fused GEMM (K=512) + bias + ReLU ----
    gemm_bf16_wmma<<<dim3(NTOK / 128, Edim / 128), 256, 0, stream>>>(
        (const __bf16*)cat_bf, (const __bf16*)fusw_bf,
        512, MODE_BIAS_RELU, fus_b, yo, nullptr, Edim);

    // ---- final LayerNorm -> d_out ----
    ln_kernel<<<NTOK, 256, 0, stream>>>(
        yo, nullptr, fus_ln_g, fus_ln_b, (float*)d_out, nullptr, 0, 0);
}
